// ContinualAttention_7541962572374
// MI455X (gfx1250) — compile-verified, matrix operand order checked by reading, not run on hardware
//
#include <hip/hip_runtime.h>

typedef _Float16 v16h  __attribute__((ext_vector_type(16)));
typedef _Float16 half8 __attribute__((ext_vector_type(8)));
typedef _Float16 half4 __attribute__((ext_vector_type(4)));
typedef float    v8f   __attribute__((ext_vector_type(8)));

#define B_     4
#define L_     2048
#define H_     8
#define D_     64
#define HD_    (H_ * D_)
#define TRAIN_ 1536
#define KSTR   72   // Ksh row stride (halfs): 144B, 16B-aligned, conflict-free rows
#define VSTR   40   // VTsh/Psh row stride (halfs): 80B, 16B-aligned
#define PSTR   40

static __device__ __forceinline__ v16h cat8(half8 a, half8 b) {
  v16h r;
#pragma unroll
  for (int i = 0; i < 8; ++i) { r[i] = a[i]; r[i + 8] = b[i]; }
  return r;
}

// Load a WMMA A-operand (16 halfs) from an LDS row in the gfx1250
// 16-bit A-layout: element e -> K = off + (e<8 ? e : 16 + (e-8)).
static __device__ __forceinline__ v16h lds_apat(const _Float16* rowp, int off) {
  const half8* p0 = (const half8*)(rowp + off);
  const half8* p1 = (const half8*)(rowp + off + 16);
  return cat8(*p0, *p1);
}

// Same pattern, loading fp32 from global (Q rows) with f16 conversion.
static __device__ __forceinline__ v16h load_q_f32(const float* rowp, int off) {
  v16h r;
#pragma unroll
  for (int e = 0; e < 8; ++e) r[e] = (_Float16)rowp[off + e];
#pragma unroll
  for (int e = 0; e < 8; ++e) r[8 + e] = (_Float16)rowp[off + 16 + e];
  return r;
}

static __device__ __forceinline__ v8f wmma16(v16h a, v16h b, v8f c) {
  return __builtin_amdgcn_wmma_f32_16x16x32_f16(false, a, false, b, (short)0, c,
                                                false, false);
}

__global__ __launch_bounds__(128)
void continual_attn_fa(const float* __restrict__ Qg,
                       const float* __restrict__ Kg,
                       const float* __restrict__ Vg,
                       const long long* __restrict__ attach,
                       float* __restrict__ Out) {
  __shared__ __align__(16) _Float16 Ksh[32 * KSTR];       // K tile, row-major f16
  __shared__ __align__(16) _Float16 VTsh[64 * VSTR];      // V tile transposed (d x keys)
  __shared__ __align__(16) _Float16 Psh[4][16 * PSTR];    // per-wave P round-trip

  const int bid  = blockIdx.x;
  const int qb   = bid & 31;                 // 32 q-blocks of 64 rows
  const int h    = (bid >> 5) & 7;
  const int b    = bid >> 8;
  const int q0b  = qb * 64;
  const int tid  = threadIdx.x;
  const int w    = tid >> 5;                 // wave id (wave32)
  const int lane = tid & 31;
  const int nlo  = lane & 15;
  const int hi   = lane >> 4;
  const int dbase = hi * 8;                  // A-layout half-wave K offset
  const int q0w  = q0b + w * 16;

  // --- Q A-operand for this wave (row m = nlo), d split into two K=32 chunks
  const float* qrow = Qg + (((size_t)b * L_ + (q0w + nlo)) * H_ + h) * D_;
  const v16h a_lo = load_q_f32(qrow, dbase);
  const v16h a_hi = load_q_f32(qrow, dbase + 32);

  // --- attach_test_after lookup per C-layout row (m = v + 8*hi)
  int att[8];
#pragma unroll
  for (int v = 0; v < 8; ++v) {
    const int q = q0w + v + 8 * hi;
    att[v] = (q >= TRAIN_) ? (int)attach[b * 64 + ((q - TRAIN_) >> 3)]
                           : 0x7fffffff;
  }

  v8f o0 = {}, o1 = {}, o2 = {}, o3 = {};
  float mrun[8], lrun[8];
#pragma unroll
  for (int v = 0; v < 8; ++v) { mrun[v] = -INFINITY; lrun[v] = 0.f; }

  const int kend = q0b + 64;  // mask is a subset of causal -> safe upper bound

  // --- software pipeline: raw fp32 tile of iteration i+1 staged in registers
  float4 kreg[4], vreg[4];
  auto stage_tile = [&](int n0) {
#pragma unroll
    for (int i = 0; i < 4; ++i) {
      const int idx = tid + 128 * i;     // 0..511 float4 chunks
      const int row = idx >> 4;          // key row 0..31
      const int seg = idx & 15;          // d segment of 4
      const size_t gbase =
          (((size_t)b * L_ + (n0 + row)) * H_ + h) * D_ + seg * 4;
      kreg[i] = *(const float4*)(Kg + gbase);
      vreg[i] = *(const float4*)(Vg + gbase);
    }
  };
  stage_tile(0);

  for (int n0 = 0; n0 < kend; n0 += 32) {
    __syncthreads();  // previous tile fully consumed by all waves
    // ---- publish staged tile: convert f32->f16, K row-major, V transposed
#pragma unroll
    for (int i = 0; i < 4; ++i) {
      const int idx = tid + 128 * i;
      const int row = idx >> 4;
      const int seg = idx & 15;
      half4 kh;
      kh[0] = (_Float16)kreg[i].x; kh[1] = (_Float16)kreg[i].y;
      kh[2] = (_Float16)kreg[i].z; kh[3] = (_Float16)kreg[i].w;
      *(half4*)&Ksh[row * KSTR + seg * 4] = kh;
      VTsh[(seg * 4 + 0) * VSTR + row] = (_Float16)vreg[i].x;
      VTsh[(seg * 4 + 1) * VSTR + row] = (_Float16)vreg[i].y;
      VTsh[(seg * 4 + 2) * VSTR + row] = (_Float16)vreg[i].z;
      VTsh[(seg * 4 + 3) * VSTR + row] = (_Float16)vreg[i].w;
    }
    __syncthreads();

    // ---- prefetch next tile's globals; latency overlaps compute below
    if (n0 + 32 < kend) stage_tile(n0 + 32);

    // ---- S = Q * K^T : two 16-key n-blocks, chained over d (2 x K=32)
    v8f c0 = {}, c1 = {};
    {
      const _Float16* kr0 = &Ksh[(0 * 16 + nlo) * KSTR];
      c0 = wmma16(a_lo, lds_apat(kr0, dbase), c0);
      c0 = wmma16(a_hi, lds_apat(kr0, dbase + 32), c0);
      const _Float16* kr1 = &Ksh[(1 * 16 + nlo) * KSTR];
      c1 = wmma16(a_lo, lds_apat(kr1, dbase), c1);
      c1 = wmma16(a_hi, lds_apat(kr1, dbase + 32), c1);
    }

    // ---- mask + online softmax (rows of a C tile live in a 16-lane group)
#pragma unroll
    for (int v = 0; v < 8; ++v) {
      const int q  = q0w + v + 8 * hi;
      const int k0 = n0 + nlo;
      const int k1 = k0 + 16;
      float s0 = c0[v] * 0.125f;   // 1/sqrt(64)
      float s1 = c1[v] * 0.125f;
      bool ok0, ok1;
      if (q < TRAIN_) {
        ok0 = (k0 <= q); ok1 = (k1 <= q);
      } else {
        const int dq = q - TRAIN_;
        ok0 = (k0 < TRAIN_) ? (k0 <= att[v])
                            : (((k0 - TRAIN_) <= dq) && (((k0 - TRAIN_) >> 3) == (dq >> 3)));
        ok1 = (k1 < TRAIN_) ? (k1 <= att[v])
                            : (((k1 - TRAIN_) <= dq) && (((k1 - TRAIN_) >> 3) == (dq >> 3)));
      }
      s0 = ok0 ? s0 : -INFINITY;
      s1 = ok1 ? s1 : -INFINITY;

      float tmax = fmaxf(s0, s1);
#pragma unroll
      for (int m = 1; m < 16; m <<= 1) tmax = fmaxf(tmax, __shfl_xor(tmax, m, 32));
      const float mold  = mrun[v];
      const float mnew  = fmaxf(mold, tmax);
      const float alpha = __expf(mold - mnew);
      const float p0    = __expf(s0 - mnew);
      const float p1    = __expf(s1 - mnew);
      float rs = p0 + p1;
#pragma unroll
      for (int m = 1; m < 16; m <<= 1) rs += __shfl_xor(rs, m, 32);
      mrun[v] = mnew;
      lrun[v] = lrun[v] * alpha + rs;
      o0[v] *= alpha; o1[v] *= alpha; o2[v] *= alpha; o3[v] *= alpha;

      // scatter P (C-layout) into per-wave LDS as f16
      const int mrow = v + 8 * hi;
      Psh[w][mrow * PSTR + nlo]      = (_Float16)p0;
      Psh[w][mrow * PSTR + nlo + 16] = (_Float16)p1;
    }
    // per-wave LDS store->load round trip; DS ops are in-order per wave
    asm volatile("s_wait_dscnt 0" ::: "memory");

    // ---- O += P * V : P re-read in A-layout, V^T rows as B-operand
    const v16h pA = lds_apat(&Psh[w][nlo * PSTR], dbase);
    o0 = wmma16(pA, lds_apat(&VTsh[(0 * 16 + nlo) * VSTR], dbase), o0);
    o1 = wmma16(pA, lds_apat(&VTsh[(1 * 16 + nlo) * VSTR], dbase), o1);
    o2 = wmma16(pA, lds_apat(&VTsh[(2 * 16 + nlo) * VSTR], dbase), o2);
    o3 = wmma16(pA, lds_apat(&VTsh[(3 * 16 + nlo) * VSTR], dbase), o3);
  }

  // ---- epilogue: normalize and store [B, L, H*D]
#pragma unroll
  for (int v = 0; v < 8; ++v) {
    const int q = q0w + v + 8 * hi;
    const float inv = 1.0f / lrun[v];
    float* orow = Out + ((size_t)b * L_ + q) * HD_ + h * D_;
    orow[0 * 16 + nlo] = o0[v] * inv;
    orow[1 * 16 + nlo] = o1[v] * inv;
    orow[2 * 16 + nlo] = o2[v] * inv;
    orow[3 * 16 + nlo] = o3[v] * inv;
  }
}

extern "C" void kernel_launch(void* const* d_in, const int* in_sizes, int n_in,
                              void* d_out, int out_size, void* d_ws, size_t ws_size,
                              hipStream_t stream) {
  (void)in_sizes; (void)n_in; (void)out_size; (void)d_ws; (void)ws_size;
  const float*     Q      = (const float*)d_in[0];
  const float*     K      = (const float*)d_in[1];
  const float*     V      = (const float*)d_in[2];
  const long long* attach = (const long long*)d_in[3];
  float* out = (float*)d_out;

  dim3 grid(B_ * H_ * (L_ / 64));   // 1024 workgroups: one (b,h,64-row q-block)
  dim3 block(128);                  // 4 waves (wave32), each owns a 16-row q tile
  hipLaunchKernelGGL(continual_attn_fa, grid, block, 0, stream, Q, K, V, attach, out);
}
